// StrokeRecoveryModel_56143812493380
// MI455X (gfx1250) — matrix operand order repaired and checked by reading, not compiled, not run on hardware
//
#include <hip/hip_runtime.h>
#include <hip/hip_bf16.h>
#include <stdint.h>

#define SEQ   256
#define BATCH 64
#define EMB   1024
#define NH    4
#define HD    256
#define NV    5

typedef __attribute__((ext_vector_type(16))) __bf16 v16bf;
typedef __attribute__((ext_vector_type(8)))  __bf16 v8bf;
typedef __attribute__((ext_vector_type(8)))  float  v8f;

// ---------- bf16 <-> f32 helpers (bit-exact RNE, no dependence on clang cast support) ----------
__device__ __forceinline__ float bf2f(__bf16 x) {
    unsigned short s; __builtin_memcpy(&s, &x, 2);
    return __uint_as_float(((unsigned)s) << 16);
}
__device__ __forceinline__ __bf16 f2bf(float x) {
    unsigned u = __float_as_uint(x);
    u += 0x7fffu + ((u >> 16) & 1u);
    unsigned short s = (unsigned short)(u >> 16);
    __bf16 o; __builtin_memcpy(&o, &s, 2);
    return o;
}

// ---------- WMMA fragment loaders (CDNA5 §7.12.2 VGPR layouts, wave32) ----------
// A: 16x32 (MxK), row-major source A[m][k], lda in elements.
// lanes 0-15 : row m0+l, K = {k0..k0+7, k0+16..k0+23}
// lanes 16-31: row m0+l, K = {k0+8..k0+15, k0+24..k0+31}
__device__ __forceinline__ v16bf load_a_frag(const __bf16* A, int lda, int m0, int k0, int lane) {
    const int half = lane >> 4;
    const __bf16* p = A + (size_t)(m0 + (lane & 15)) * lda + k0 + half * 8;
    union { v16bf v; v8bf h[2]; } u;
    u.h[0] = *(const v8bf*)p;
    u.h[1] = *(const v8bf*)(p + 16);
    return u.v;
}
// B: 32x16 (KxN) with B[k][n] = W[n][k], W row-major [out,in], ldw in elements.
// lanes 0-15 : col n0+l, K = {k0..k0+7, k0+16..k0+23}
// lanes 16-31: col n0+l, K = {k0+8..k0+15, k0+24..k0+31}
__device__ __forceinline__ v16bf load_b_frag(const __bf16* W, int ldw, int n0, int k0, int lane) {
    const int half = lane >> 4;
    const __bf16* p = W + (size_t)(n0 + (lane & 15)) * ldw + k0 + half * 8;
    union { v16bf v; v8bf h[2]; } u;
    u.h[0] = *(const v8bf*)p;
    u.h[1] = *(const v8bf*)(p + 16);
    return u.v;
}
__device__ __forceinline__ v8f wmma_bf16(v16bf a, v16bf b, v8f c) {
    return __builtin_amdgcn_wmma_f32_16x16x32_bf16(false, a, false, b, (short)0, c, false, false);
}

// ---------- one-time conversion / init ----------
__global__ void cvt_bf16_kernel(const float* __restrict__ src, __bf16* __restrict__ dst, int n) {
    const int stride = gridDim.x * blockDim.x;
    for (int i = blockIdx.x * blockDim.x + threadIdx.x; i < n; i += stride)
        dst[i] = f2bf(src[i]);
}

__global__ void zero_state_kernel(float* c1, float* c2, __bf16* h1, __bf16* h2,
                                  float* h2f, __bf16* ctx) {
    const int stride = gridDim.x * blockDim.x;
    const int NE = BATCH * EMB;
    const __bf16 zb = f2bf(0.0f);
    for (int i = blockIdx.x * blockDim.x + threadIdx.x; i < NE; i += stride) {
        c1[i] = 0.f; c2[i] = 0.f;
        h2f[i] = 0.f; h2f[i + NE] = 0.f;
        h1[i] = zb;  h1[i + NE] = zb;
        h2[i] = zb;  h2[i + NE] = zb;
        ctx[i] = zb;
    }
}

// ---------- K/V projection: [S*B,1024] @ Wk^T / Wv^T (bf16 WMMA, f32 accumulate) ----------
__global__ __launch_bounds__(128)
void kv_proj_kernel(const __bf16* __restrict__ cnn_bf, const __bf16* __restrict__ wqkv_bf,
                    const float* __restrict__ bqkv, __bf16* __restrict__ Kt,
                    __bf16* __restrict__ Vt) {
    const int mt = blockIdx.x;            // 0..(S*B/16-1)
    const int ng = blockIdx.y;            // 0..15
    const int z  = blockIdx.z;            // 0=K, 1=V
    const int wave = threadIdx.x >> 5, lane = threadIdx.x & 31;
    const int half = lane >> 4, lcol = lane & 15;
    const int n0 = (ng * 4 + wave) * 16;
    const int m0 = mt * 16;
    const __bf16* W = wqkv_bf + (size_t)(1 + z) * EMB * EMB;   // k rows then v rows
    __bf16* Out = z ? Vt : Kt;
    v8f acc = {};
    for (int k0 = 0; k0 < EMB; k0 += 32) {
        v16bf a = load_a_frag(cnn_bf, EMB, m0, k0, lane);
        v16bf b = load_b_frag(W, EMB, n0, k0, lane);
        acc = wmma_bf16(a, b, acc);
    }
    const int col = n0 + lcol;
    const float bias = bqkv[(1 + z) * EMB + col];
#pragma unroll
    for (int j = 0; j < 8; ++j)
        Out[(size_t)(m0 + half * 8 + j) * EMB + col] = f2bf(acc[j] + bias);
}

// ---------- per-step attention: q-proj (WMMA) -> softmax attention (VALU) -> out-proj (WMMA) ----------
__global__ __launch_bounds__(512)
void attn_step_kernel(const __bf16* __restrict__ r,      // [64,1024] = prev h2 (bf16)
                      const __bf16* __restrict__ wq,     // rows 0..1023 of w_qkv (bf16)
                      const float*  __restrict__ bqkv,   // q bias = first 1024
                      const __bf16* __restrict__ Kt,     // [S*B,1024]
                      const __bf16* __restrict__ Vt,     // [S*B,1024]
                      const __bf16* __restrict__ wo,     // [1024,1024]
                      const float*  __restrict__ bo,
                      __bf16* __restrict__ ctx_out) {    // [64,1024]
    extern __shared__ __bf16 smem[];
    __bf16* q_sh   = smem;               // [16][1024] 32KB
    __bf16* ctx_sh = smem + 16 * EMB;    // [16][1024] 32KB

    const int g = blockIdx.x;            // batch group (4 groups of 16 rows)
    const int m0 = g * 16;
    const int wave = threadIdx.x >> 5, lane = threadIdx.x & 31;
    const int half = lane >> 4, lcol = lane & 15;

    // Stage A: q = r @ wq^T + bq  -> q_sh (bf16)
    for (int nt = wave; nt < EMB / 16; nt += 16) {
        const int n0 = nt * 16;
        v8f acc = {};
        for (int k0 = 0; k0 < EMB; k0 += 32) {
            v16bf a = load_a_frag(r, EMB, m0, k0, lane);
            v16bf b = load_b_frag(wq, EMB, n0, k0, lane);
            acc = wmma_bf16(a, b, acc);
        }
        const int col = n0 + lcol;
        const float bias = bqkv[col];
#pragma unroll
        for (int j = 0; j < 8; ++j)
            q_sh[(half * 8 + j) * EMB + col] = f2bf(acc[j] + bias);
    }
    __syncthreads();

    // Stage B: wave w owns local batch row w; loop over 4 heads
    const int bl = wave;
    const int bg = m0 + bl;
    const float scale = 0.0625f;  // 1/sqrt(256)
    for (int h = 0; h < NH; ++h) {
        const __bf16* qrow = q_sh + bl * EMB + h * HD;
        float sc[8];
#pragma unroll
        for (int j = 0; j < 8; ++j) {                 // lane owns s = j*32 + lane
            const int s = j * 32 + lane;
            const __bf16* krow = Kt + ((size_t)s * BATCH + bg) * EMB + h * HD;
            float acc = 0.f;
            for (int d = 0; d < HD; d += 8) {
                v8bf qv = *(const v8bf*)(qrow + d);   // LDS broadcast
                v8bf kv = *(const v8bf*)(krow + d);   // L2-resident K
#pragma unroll
                for (int e = 0; e < 8; ++e) acc += bf2f(qv[e]) * bf2f(kv[e]);
            }
            sc[j] = acc * scale;
        }
        float mx = sc[0];
#pragma unroll
        for (int j = 1; j < 8; ++j) mx = fmaxf(mx, sc[j]);
        for (int o = 16; o > 0; o >>= 1) mx = fmaxf(mx, __shfl_xor(mx, o, 32));
        float sum = 0.f;
#pragma unroll
        for (int j = 0; j < 8; ++j) { sc[j] = __expf(sc[j] - mx); sum += sc[j]; }
        for (int o = 16; o > 0; o >>= 1) sum += __shfl_xor(sum, o, 32);
        const float inv = 1.f / sum;
#pragma unroll
        for (int j = 0; j < 8; ++j) sc[j] *= inv;

        // ctx: lane owns d = lane*8..+7 ; attn weights broadcast via shuffles
        const int d0 = lane * 8;
        float cacc[8] = {0.f, 0.f, 0.f, 0.f, 0.f, 0.f, 0.f, 0.f};
        for (int j = 0; j < 8; ++j) {
            const float pj = sc[j];
            for (int sl = 0; sl < 32; ++sl) {
                const float a = __shfl(pj, sl, 32);
                const int s = j * 32 + sl;
                const __bf16* vrow = Vt + ((size_t)s * BATCH + bg) * EMB + h * HD + d0;
                v8bf vv = *(const v8bf*)vrow;
#pragma unroll
                for (int e = 0; e < 8; ++e) cacc[e] += a * bf2f(vv[e]);
            }
        }
#pragma unroll
        for (int e = 0; e < 8; ++e)
            ctx_sh[bl * EMB + h * HD + d0 + e] = f2bf(cacc[e]);
    }
    __syncthreads();

    // Stage C: ctx @ wo^T + bo -> ctx_out
    for (int nt = wave; nt < EMB / 16; nt += 16) {
        const int n0 = nt * 16;
        v8f acc = {};
        for (int k0 = 0; k0 < EMB; k0 += 32) {
            v16bf a = load_a_frag(ctx_sh, EMB, 0, k0, lane);
            v16bf b = load_b_frag(wo, EMB, n0, k0, lane);
            acc = wmma_bf16(a, b, acc);
        }
        const int col = n0 + lcol;
        const float bias = bo[col];
#pragma unroll
        for (int j = 0; j < 8; ++j)
            ctx_out[(size_t)(m0 + half * 8 + j) * EMB + col] = f2bf(acc[j] + bias);
    }
}

// ---------- LSTM cell: one block = one 16x16 tile of all 4 gates (wave per gate), fused update ----------
__global__ __launch_bounds__(128)
void lstm_step_kernel(const __bf16* __restrict__ xa,          // [64,1024]
                      const __bf16* __restrict__ xb,          // [64,1024] or null
                      const __bf16* __restrict__ wih, int ldih, // [4096, ldih]
                      const __bf16* __restrict__ whh,         // [4096,1024]
                      const float* __restrict__ bih, const float* __restrict__ bhh,
                      const __bf16* __restrict__ hprev,       // [64,1024]
                      float* __restrict__ c,                  // [64,1024] in/out (elementwise safe)
                      __bf16* __restrict__ hout,              // [64,1024]
                      float* __restrict__ hout_f32) {         // optional
    __shared__ float gsh[4][16][16];
    const int jt = blockIdx.x;          // 0..63 hidden-column tile
    const int mt = blockIdx.y;          // 0..3  batch tile
    const int m0 = mt * 16;
    const int wave = threadIdx.x >> 5, lane = threadIdx.x & 31;
    const int half = lane >> 4, lcol = lane & 15;
    const int n0 = wave * EMB + jt * 16;   // gate wave selects i/f/g/o row block

    v8f acc = {};
    for (int k0 = 0; k0 < EMB; k0 += 32) {
        v16bf a = load_a_frag(xa, EMB, m0, k0, lane);
        v16bf b = load_b_frag(wih, ldih, n0, k0, lane);
        acc = wmma_bf16(a, b, acc);
    }
    if (xb) {  // second half of concatenated x (LSTM1 only): w_ih columns 1024..2047
        for (int k0 = 0; k0 < EMB; k0 += 32) {
            v16bf a = load_a_frag(xb, EMB, m0, k0, lane);
            v16bf b = load_b_frag(wih, ldih, n0, EMB + k0, lane);
            acc = wmma_bf16(a, b, acc);
        }
    }
    for (int k0 = 0; k0 < EMB; k0 += 32) {
        v16bf a = load_a_frag(hprev, EMB, m0, k0, lane);
        v16bf b = load_b_frag(whh, EMB, n0, k0, lane);
        acc = wmma_bf16(a, b, acc);
    }
    const float bias = bih[n0 + lcol] + bhh[n0 + lcol];
#pragma unroll
    for (int j = 0; j < 8; ++j)
        gsh[wave][half * 8 + j][lcol] = acc[j] + bias;
    __syncthreads();

    for (int e = threadIdx.x; e < 256; e += 128) {
        const int rl = e >> 4, cl = e & 15;
        const int b = m0 + rl;
        const int jc = jt * 16 + cl;
        const float gi = gsh[0][rl][cl];
        const float gf = gsh[1][rl][cl];
        const float gg = gsh[2][rl][cl];
        const float go = gsh[3][rl][cl];
        const float si = 1.f / (1.f + __expf(-gi));
        const float sf = 1.f / (1.f + __expf(-gf));
        const float so = 1.f / (1.f + __expf(-go));
        const size_t idx = (size_t)b * EMB + jc;
        const float cn = sf * c[idx] + si * tanhf(gg);
        const float hn = so * tanhf(cn);
        c[idx] = cn;
        hout[idx] = f2bf(hn);
        if (hout_f32) hout_f32[idx] = hn;
    }
}

// ---------- output head: 64x5 dots of 1024 + partial sigmoid ----------
__global__ __launch_bounds__(320)
void head_kernel(const float* __restrict__ h2, const float* __restrict__ wout,
                 const float* __restrict__ bout, float* __restrict__ out) {
    const int t = threadIdx.x;           // 320 = 64*5
    const int b = t / NV, v = t % NV;
    float acc = bout[v];
    const float* hr = h2 + (size_t)b * EMB;
    const float* wr = wout + (size_t)v * EMB;
    for (int k = 0; k < EMB; ++k) acc = fmaf(hr[k], wr[k], acc);
    if (v >= 2) acc = 1.f / (1.f + __expf(-acc));
    out[b * NV + v] = acc;
}

// =====================================================================================
extern "C" void kernel_launch(void* const* d_in, const int* in_sizes, int n_in,
                              void* d_out, int out_size, void* d_ws, size_t ws_size,
                              hipStream_t stream) {
    const float* cnn  = (const float*)d_in[0];
    const float* wqkv = (const float*)d_in[1];
    const float* bqkv = (const float*)d_in[2];
    const float* wo   = (const float*)d_in[3];
    const float* bo   = (const float*)d_in[4];
    const float* wih1 = (const float*)d_in[5];
    const float* whh1 = (const float*)d_in[6];
    const float* bih1 = (const float*)d_in[7];
    const float* bhh1 = (const float*)d_in[8];
    const float* wih2 = (const float*)d_in[9];
    const float* whh2 = (const float*)d_in[10];
    const float* bih2 = (const float*)d_in[11];
    const float* bhh2 = (const float*)d_in[12];
    const float* wout = (const float*)d_in[13];
    const float* bout = (const float*)d_in[14];

    char* ws = (char*)d_ws;
    size_t off = 0;
    auto carve = [&](size_t bytes) -> void* {
        void* p = ws + off;
        off = (off + bytes + 255) & ~(size_t)255;
        return p;
    };
    __bf16* wqkv_bf = (__bf16*)carve((size_t)3 * EMB * EMB * 2);
    __bf16* wo_bf   = (__bf16*)carve((size_t)EMB * EMB * 2);
    __bf16* wih1_bf = (__bf16*)carve((size_t)4 * EMB * 2 * EMB * 2);
    __bf16* whh1_bf = (__bf16*)carve((size_t)4 * EMB * EMB * 2);
    __bf16* wih2_bf = (__bf16*)carve((size_t)4 * EMB * EMB * 2);
    __bf16* whh2_bf = (__bf16*)carve((size_t)4 * EMB * EMB * 2);
    __bf16* cnn_bf  = (__bf16*)carve((size_t)SEQ * BATCH * EMB * 2);
    __bf16* K_bf    = (__bf16*)carve((size_t)SEQ * BATCH * EMB * 2);
    __bf16* V_bf    = (__bf16*)carve((size_t)SEQ * BATCH * EMB * 2);
    __bf16* ctx_bf  = (__bf16*)carve((size_t)BATCH * EMB * 2);
    __bf16* h1_bf   = (__bf16*)carve((size_t)2 * BATCH * EMB * 2);   // ping-pong
    __bf16* h2_bf   = (__bf16*)carve((size_t)2 * BATCH * EMB * 2);   // ping-pong
    float*  h2_f    = (float*)carve((size_t)2 * BATCH * EMB * 4);    // ping-pong
    float*  c1      = (float*)carve((size_t)BATCH * EMB * 4);
    float*  c2      = (float*)carve((size_t)BATCH * EMB * 4);

    // One-time conversions + state init (deterministic each call)
    cvt_bf16_kernel<<<2048, 256, 0, stream>>>(cnn,  cnn_bf,  SEQ * BATCH * EMB);
    cvt_bf16_kernel<<<1024, 256, 0, stream>>>(wqkv, wqkv_bf, 3 * EMB * EMB);
    cvt_bf16_kernel<<<512,  256, 0, stream>>>(wo,   wo_bf,   EMB * EMB);
    cvt_bf16_kernel<<<2048, 256, 0, stream>>>(wih1, wih1_bf, 4 * EMB * 2 * EMB);
    cvt_bf16_kernel<<<1024, 256, 0, stream>>>(whh1, whh1_bf, 4 * EMB * EMB);
    cvt_bf16_kernel<<<1024, 256, 0, stream>>>(wih2, wih2_bf, 4 * EMB * EMB);
    cvt_bf16_kernel<<<1024, 256, 0, stream>>>(whh2, whh2_bf, 4 * EMB * EMB);
    zero_state_kernel<<<64, 256, 0, stream>>>(c1, c2, h1_bf, h2_bf, h2_f, ctx_bf);

    // K/V projections (precomputed once, bf16, L2-resident for the whole scan)
    kv_proj_kernel<<<dim3(SEQ * BATCH / 16, 16, 2), 128, 0, stream>>>(cnn_bf, wqkv_bf, bqkv,
                                                                      K_bf, V_bf);

    const size_t NE = (size_t)BATCH * EMB;
    for (int s = 0; s < SEQ; ++s) {
        const int cur = s & 1, nxt = cur ^ 1;
        attn_step_kernel<<<4, 512, 2 * 16 * EMB * 2, stream>>>(
            h2_bf + cur * NE, wqkv_bf, bqkv, K_bf, V_bf, wo_bf, bo, ctx_bf);
        lstm_step_kernel<<<dim3(64, 4), 128, 0, stream>>>(
            h2_bf + cur * NE, ctx_bf, wih1_bf, 2 * EMB, whh1_bf, bih1, bhh1,
            h1_bf + cur * NE, c1, h1_bf + nxt * NE, nullptr);
        lstm_step_kernel<<<dim3(64, 4), 128, 0, stream>>>(
            h1_bf + nxt * NE, nullptr, wih2_bf, EMB, whh2_bf, bih2, bhh2,
            h2_bf + cur * NE, c2, h2_bf + nxt * NE, h2_f + nxt * NE);
        head_kernel<<<1, 320, 0, stream>>>(
            h2_f + nxt * NE, wout, bout, (float*)d_out + (size_t)s * BATCH * NV);
    }
    (void)in_sizes; (void)n_in; (void)out_size; (void)ws_size;
}
